// GUIDE_7086696038457
// MI455X (gfx1250) — matrix operand order repaired and checked by reading, not compiled, not run on hardware
//
#include <hip/hip_runtime.h>
#include <hip/hip_bf16.h>
#include <stdint.h>

typedef __attribute__((ext_vector_type(16))) _Float16 v16h;
typedef __attribute__((ext_vector_type(8)))  float    v8f;

#define BB  4
#define HH  128
#define WWD 128
#define HWX (HH*WWD)
#define HP  (HH+2)     // padded height
#define WP  (WWD+2)    // padded width

union Frag {
  v16h h;
  uint4 q[2];
};

// ---------------------------------------------------------------- utilities
__global__ void zero_f32(float* p, int n) {
  int i = blockIdx.x * blockDim.x + threadIdx.x;
  if (i < n) p[i] = 0.f;
}

// Zero the 1-pixel halo of a padded NHWC f16 buffer [B, H+2, W+2, C].
// 516 border pixels per batch; 8 halves (uint4) per thread.
__global__ void zero_halo_f16(_Float16* __restrict__ buf, int C) {
  int i = blockIdx.x * blockDim.x + threadIdx.x;
  int grp = C >> 3;
  int tot = BB * 516 * grp;
  if (i >= tot) return;
  int g = i % grp;
  int r = i / grp;
  int b = r / 516;
  int pix = r - b * 516;
  int h, w;
  if (pix < WP)          { h = 0;      w = pix; }
  else if (pix < 2 * WP) { h = HP - 1; w = pix - WP; }
  else { int t = pix - 2 * WP; h = 1 + (t >> 1); w = (t & 1) ? (WP - 1) : 0; }
  uint4* p = (uint4*)(buf + (((size_t)b * HP + h) * WP + w) * C) + g;
  *p = make_uint4(0, 0, 0, 0);
}

// Repack OIHW f32 conv weight -> [N][t*Cin + c] f16 (N-major, K contiguous)
__global__ void pack_w3x3(const float* __restrict__ w, _Float16* __restrict__ wpk,
                          int Cout, int Cin) {
  int i = blockIdx.x * blockDim.x + threadIdx.x;
  int tot = Cout * Cin * 9;
  if (i >= tot) return;
  int n = i / (Cin * 9);
  int r = i - n * (Cin * 9);
  int t = r / Cin;
  int c = r - t * Cin;
  wpk[i] = (_Float16)w[((size_t)(n * Cin + c)) * 9 + t];
}

// cat = concat(input, weight) as f16 padded NHWC [B,H+2,W+2,128] (interior)
__global__ void build_cat(const float* __restrict__ in, const float* __restrict__ wt,
                          _Float16* __restrict__ cat) {
  size_t i = (size_t)blockIdx.x * blockDim.x + threadIdx.x;
  size_t tot = (size_t)BB * 128 * HWX;
  if (i >= tot) return;
  int b = (int)(i / (128 * HWX));
  int r = (int)(i - (size_t)b * (128 * HWX));
  int c = r / HWX;
  int p = r - c * HWX;
  int h = p >> 7, w = p & 127;
  float v = (c < 64) ? in[((size_t)(b * 64 + c)) * HWX + p]
                     : wt[((size_t)(b * 64 + c - 64)) * HWX + p];
  cat[(((size_t)b * HP + h + 1) * WP + w + 1) * 128 + c] = (_Float16)v;
}

// -------------------------------------------------- implicit GEMM via WMMA
// Y[b,h,w,n] = act( sum_{taps,c} X[b,h+dy,w+dx,c] * Wpk[n][tap*Cin+c] + bias[n] )
// Wave tile: 16 pixels (along W) x 16 out-channels; 8 waves cover one W row.
// 3x3 inputs are halo-padded NHWC => all A loads unconditional b128;
// weights staged to LDS => B frags are ds_load_b128. Inner body is pure
// loads + v_wmma (no boundary VALU, no exec masking).
template <int CIN, int NTAPS, bool HPAD, bool HAS_BIAS, bool RELU,
          bool WRITE_H, bool WRITE_F, bool F_NCHW, bool DO_STATS>
__launch_bounds__(256)
__global__ void conv_gemm_wmma(const _Float16* __restrict__ X,
                               const _Float16* __restrict__ Wpk, int wstride,
                               const float* __restrict__ bias,
                               _Float16* __restrict__ Yh,       // NHWC f16
                               float* __restrict__ Yf,          // f32 out
                               float* __restrict__ stats,       // [2*Cout]
                               int Cout) {
  constexpr int  KT   = NTAPS * CIN;
  constexpr bool IPAD = (NTAPS == 9);      // padded input layout
  __shared__ __align__(16) _Float16 sB[16 * KT];
  const int row = blockIdx.x;              // b*H + h
  const int b   = row / HH;
  const int h   = row - b * HH;
  const int n0  = blockIdx.y * 16;
  const int tid = threadIdx.x;

  { // stage 16 weight rows [n0..n0+15][0..KT) into LDS
    const unsigned* wsrc = (const unsigned*)(Wpk + (size_t)b * wstride);
    unsigned* sB32 = (unsigned*)sB;
    constexpr int KW2 = KT >> 1;
    for (int i = tid; i < 16 * KW2; i += 256) {
      int ni = i / KW2;
      int kk = i - ni * KW2;
      sB32[i] = wsrc[(size_t)(n0 + ni) * KW2 + kk];
    }
  }
  __syncthreads();

  const int wave  = tid >> 5;
  const int lane  = tid & 31;
  const int mrow  = lane & 15;    // A-matrix row this lane feeds
  const int khalf = lane >> 4;
  const int nloc  = lane & 15;    // D-matrix column this lane owns
  const int w0    = wave * 16;    // 8 waves cover one W=128 row

  v8f acc = {};
#pragma unroll
  for (int tap = 0; tap < NTAPS; ++tap) {
    const int dy = (NTAPS == 9) ? (tap / 3) : 0;       // 0..2 into padded rows
    const int dx = (NTAPS == 9) ? (tap % 3) : 0;
    const _Float16* xp;
    if (IPAD)
      xp = X + (((size_t)b * HP + h + dy) * WP + (w0 + mrow + dx)) * CIN + khalf * 8;
    else
      xp = X + (((size_t)b * HH + h) * WWD + (w0 + mrow)) * CIN + khalf * 8;
    const _Float16* bp = sB + (size_t)nloc * KT + tap * CIN + khalf * 8;
#pragma unroll
    for (int cc = 0; cc < CIN / 32; ++cc) {
      Frag a, bf;
      // A frag 16x32 f16: lanes 0-15 -> K {0..7,16..23}; 16-31 -> K {8..15,24..31}
      a.q[0] = *(const uint4*)xp;
      a.q[1] = *(const uint4*)(xp + 16);
      // B frag 32x16 f16: lanes 0-15 -> K {0..7,16..23}; 16-31 -> K {8..15,24..31} for column nloc
      bf.q[0] = *(const uint4*)bp;
      bf.q[1] = *(const uint4*)(bp + 16);
      acc = __builtin_amdgcn_wmma_f32_16x16x32_f16(false, a.h, false, bf.h,
                                                   (short)0, acc, false, false);
      xp += 32;
      bp += 32;
    }
  }

  const int n = n0 + nloc;
  const float bv = HAS_BIAS ? bias[n] : 0.f;
  float lsum = 0.f, lsq = 0.f;
#pragma unroll
  for (int i = 0; i < 8; ++i) {
    const int m = khalf * 8 + i;            // D row -> pixel w0+m
    float v = acc[i] + bv;
    if (RELU) v = v > 0.f ? v : 0.f;
    lsum += v;
    lsq  += v * v;
    const int wp = w0 + m;
    if (WRITE_H) {
      size_t oi;
      if (HPAD) oi = (((size_t)b * HP + h + 1) * WP + wp + 1) * (size_t)Cout + n;
      else      oi = (((size_t)b * HH + h) * WWD + wp) * (size_t)Cout + n;
      Yh[oi] = (_Float16)v;
    }
    if (WRITE_F) {
      if (F_NCHW) Yf[(((size_t)b * Cout + n) * HH + h) * WWD + wp] = v;
      else        Yf[(((size_t)b * HH + h) * WWD + wp) * (size_t)Cout + n] = v;
    }
  }
  if (DO_STATS) {
    atomicAdd(&stats[n], lsum);
    atomicAdd(&stats[Cout + n], lsq);
  }
}

// ---------------------------------------------------------------- pooling
__global__ void pool_mean(const _Float16* __restrict__ w21h, float* __restrict__ pooled) {
  __shared__ float red[256];
  const int b = blockIdx.x >> 6;
  const int c = blockIdx.x & 63;
  float s = 0.f;
  for (int p = threadIdx.x; p < HWX; p += 256)
    s += (float)w21h[((size_t)b * HWX + p) * 64 + c];
  red[threadIdx.x] = s;
  __syncthreads();
  for (int st = 128; st > 0; st >>= 1) {
    if (threadIdx.x < st) red[threadIdx.x] += red[threadIdx.x + st];
    __syncthreads();
  }
  if (threadIdx.x == 0) pooled[blockIdx.x] = red[0] / (float)HWX;
}

// w22[b,i,j] f16 = b22[i*64+j] + sum_c wc22[(i*64+j)*64+c] * pooled[b,c]
__global__ void make_w22(const float* __restrict__ wc22, const float* __restrict__ bc22,
                         const float* __restrict__ pooled, _Float16* __restrict__ w22h) {
  int i = blockIdx.x * blockDim.x + threadIdx.x;
  if (i >= BB * 4096) return;
  int b  = i >> 12;
  int ij = i & 4095;
  float s = bc22[ij];
  const float* pr = pooled + b * 64;
  const float* wr = wc22 + (size_t)ij * 64;
#pragma unroll 8
  for (int c = 0; c < 64; ++c) s += wr[c] * pr[c];
  w22h[i] = (_Float16)s;
}

// per-pixel depthwise 3x3 with predicted weights + BN stat accumulation
__global__ void local_conv_bnstat(const float* __restrict__ in, const _Float16* __restrict__ w12h,
                                  float* __restrict__ lc, float* __restrict__ stats) {
  __shared__ float ssum[64], ssq[64];
  if (threadIdx.x < 64) { ssum[threadIdx.x] = 0.f; ssq[threadIdx.x] = 0.f; }
  __syncthreads();
  const int ci = threadIdx.x & 63;
  const int pj = threadIdx.x >> 6;
  const size_t gp = (size_t)blockIdx.x * 4 + pj;   // pixel across B*HW
  const int b = (int)(gp / HWX);
  const int p = (int)(gp - (size_t)b * HWX);
  const int h = p >> 7;
  const int w = p & 127;
  const float* inp = in + ((size_t)b * 64 + ci) * HWX;
  const _Float16* wp = w12h + ((size_t)b * HWX + p) * 576 + ci * 9;
  float s = 0.f;
#pragma unroll
  for (int t = 0; t < 9; ++t) {
    const int dy = t / 3 - 1, dx = t % 3 - 1;
    const int hh = h + dy, ww = w + dx;
    const float xv = (hh >= 0 && hh < HH && ww >= 0 && ww < WWD) ? inp[hh * WWD + ww] : 0.f;
    s += xv * (float)wp[t];
  }
  lc[((size_t)b * HWX + p) * 64 + ci] = s;
  atomicAdd(&ssum[ci], s);
  atomicAdd(&ssq[ci], s * s);
  __syncthreads();
  if (threadIdx.x < 64) {
    atomicAdd(&stats[threadIdx.x], ssum[threadIdx.x]);
    atomicAdd(&stats[64 + threadIdx.x], ssq[threadIdx.x]);
  }
}

// BN(+relu) of f32 NHWC input -> f16 NHWC output; OPAD selects halo-padded out.
template <bool OPAD>
__global__ void bn_relu_nhwc_f16(const float* __restrict__ x, const float* __restrict__ stats,
                                 const float* __restrict__ g, const float* __restrict__ bt,
                                 _Float16* __restrict__ out) {
  size_t i = (size_t)blockIdx.x * blockDim.x + threadIdx.x;
  if (i >= (size_t)BB * HWX * 64) return;
  const int c = (int)(i & 63);
  const float cnt = (float)(BB * HWX);
  const float m = stats[c] / cnt;
  const float v = stats[64 + c] / cnt - m * m;
  const float y = g[c] * (x[i] - m) * rsqrtf(v + 1e-5f) + bt[c];
  const _Float16 r = (_Float16)(y > 0.f ? y : 0.f);
  if (OPAD) {
    size_t pix = i >> 6;
    int b = (int)(pix / HWX);
    int p = (int)(pix - (size_t)b * HWX);
    int h = p >> 7, w = p & 127;
    out[(((size_t)b * HP + h + 1) * WP + w + 1) * 64 + c] = r;
  } else {
    out[i] = r;
  }
}

__global__ void bn_relu_nchw_f32(const float* __restrict__ x, const float* __restrict__ stats,
                                 const float* __restrict__ g, const float* __restrict__ bt,
                                 float* __restrict__ out) {
  size_t i = (size_t)blockIdx.x * blockDim.x + threadIdx.x;
  if (i >= (size_t)BB * 64 * HWX) return;
  const int c = (int)((i / HWX) & 63);
  const float cnt = (float)(BB * HWX);
  const float m = stats[c] / cnt;
  const float v = stats[64 + c] / cnt - m * m;
  const float y = g[c] * (x[i] - m) * rsqrtf(v + 1e-5f) + bt[c];
  out[i] = y > 0.f ? y : 0.f;
}

// ---------------------------------------------------------------- driver
extern "C" void kernel_launch(void* const* d_in, const int* in_sizes, int n_in,
                              void* d_out, int out_size, void* d_ws, size_t ws_size,
                              hipStream_t stream) {
  (void)in_sizes; (void)n_in; (void)out_size; (void)ws_size;
  const float* in   = (const float*)d_in[0];
  const float* wt   = (const float*)d_in[1];
  const float* w11w = (const float*)d_in[2];
  const float* b11  = (const float*)d_in[3];
  const float* w12w = (const float*)d_in[4];
  const float* b12  = (const float*)d_in[5];
  const float* w21w = (const float*)d_in[6];
  const float* b21  = (const float*)d_in[7];
  const float* w22w = (const float*)d_in[8];
  const float* b22  = (const float*)d_in[9];
  const float* lbg  = (const float*)d_in[10];
  const float* lbb  = (const float*)d_in[11];
  const float* brg  = (const float*)d_in[12];
  const float* brb  = (const float*)d_in[13];
  const float* w3w  = (const float*)d_in[14];
  const float* c3g  = (const float*)d_in[15];
  const float* c3b  = (const float*)d_in[16];
  float* out = (float*)d_out;

  char* ws = (char*)d_ws;
  size_t off = 0;
  auto carve = [&](size_t bytes) -> char* {
    char* p = ws + off;
    off += (bytes + 255) & ~(size_t)255;
    return p;
  };
  const size_t PADPIX = (size_t)BB * HP * WP;                      // padded pixels
  _Float16* catb   = (_Float16*)carve(PADPIX * 128 * 2);           // 17.3 MB padded
  _Float16* wpk11  = (_Float16*)carve((size_t)64 * 1152 * 2);
  _Float16* wpk21  = (_Float16*)carve((size_t)64 * 1152 * 2);
  _Float16* wpk12  = (_Float16*)carve((size_t)576 * 576 * 2);
  _Float16* wpk3   = (_Float16*)carve((size_t)64 * 576 * 2);
  _Float16* w11h   = (_Float16*)carve(PADPIX * 64 * 2);            // padded
  _Float16* w21h   = (_Float16*)carve((size_t)BB * HWX * 64 * 2);  // unpadded
  _Float16* w12h   = (_Float16*)carve((size_t)BB * HWX * 576 * 2); // 75.5 MB
  float*    pooled = (float*)carve((size_t)BB * 64 * 4);
  _Float16* w22h   = (_Float16*)carve((size_t)BB * 4096 * 2);
  float*    lc     = (float*)carve((size_t)BB * HWX * 64 * 4);     // 16.8 MB
  _Float16* out1h  = (_Float16*)carve((size_t)BB * HWX * 64 * 2);  // unpadded
  float*    out2f  = (float*)carve((size_t)BB * HWX * 64 * 4);
  _Float16* out3h  = (_Float16*)carve(PADPIX * 64 * 2);            // padded
  float*    c3raw  = (float*)carve((size_t)BB * 64 * HWX * 4);
  float*    stats  = (float*)carve(3 * 128 * 4);
  float* stats0 = stats;          // local-conv BN
  float* stats1 = stats + 128;    // out2 BN
  float* stats2 = stats + 256;    // conv3 BN

  // 0) zero BN accumulators + halos of padded WMMA inputs
  zero_f32<<<2, 256, 0, stream>>>(stats, 384);
  zero_halo_f16<<<(BB * 516 * (128 / 8) + 255) / 256, 256, 0, stream>>>(catb, 128);
  zero_halo_f16<<<(BB * 516 * (64 / 8) + 255) / 256, 256, 0, stream>>>(w11h, 64);
  zero_halo_f16<<<(BB * 516 * (64 / 8) + 255) / 256, 256, 0, stream>>>(out3h, 64);
  // 1) repack conv weights to [N][K] f16
  pack_w3x3<<<(64 * 128 * 9 + 255) / 256, 256, 0, stream>>>(w11w, wpk11, 64, 128);
  pack_w3x3<<<(64 * 128 * 9 + 255) / 256, 256, 0, stream>>>(w21w, wpk21, 64, 128);
  pack_w3x3<<<(576 * 64 * 9 + 255) / 256, 256, 0, stream>>>(w12w, wpk12, 576, 64);
  pack_w3x3<<<(64 * 64 * 9 + 255) / 256, 256, 0, stream>>>(w3w, wpk3, 64, 64);
  // 2) cat -> f16 padded NHWC
  build_cat<<<(BB * 128 * HWX + 255) / 256, 256, 0, stream>>>(in, wt, catb);
  // 3) w11 = relu(conv11(cat)) (padded out), w21 = relu(conv21(cat))   [WMMA]
  conv_gemm_wmma<128, 9, true, true, true, true, false, false, false>
      <<<dim3(BB * HH, 4), 256, 0, stream>>>(catb, wpk11, 0, b11, w11h, nullptr, nullptr, 64);
  conv_gemm_wmma<128, 9, false, true, true, true, false, false, false>
      <<<dim3(BB * HH, 4), 256, 0, stream>>>(catb, wpk21, 0, b21, w21h, nullptr, nullptr, 64);
  // 4) w12 = conv12(w11)   [WMMA, Cout=576]
  conv_gemm_wmma<64, 9, false, true, false, true, false, false, false>
      <<<dim3(BB * HH, 36), 256, 0, stream>>>(w11h, wpk12, 0, b12, w12h, nullptr, nullptr, 576);
  // 5) pooled = mean(w21); w22 = conv22(pooled)
  pool_mean<<<BB * 64, 256, 0, stream>>>(w21h, pooled);
  make_w22<<<(BB * 4096 + 255) / 256, 256, 0, stream>>>(w22w, b22, pooled, w22h);
  // 6) lc = local_conv(input, w12) + BN stats; out1 = relu(bn(lc))
  local_conv_bnstat<<<BB * HWX / 4, 256, 0, stream>>>(in, w12h, lc, stats0);
  bn_relu_nhwc_f16<false><<<(BB * HWX * 64 + 255) / 256, 256, 0, stream>>>(lc, stats0, lbg, lbb, out1h);
  // 7) out2 = w22 @ out1 (per-batch 64x64 x pixels)   [WMMA, ntaps=1]
  conv_gemm_wmma<64, 1, false, false, false, false, true, false, true>
      <<<dim3(BB * HH, 4), 256, 0, stream>>>(out1h, w22h, 4096, nullptr, nullptr, out2f, stats1, 64);
  // 8) out3 = relu(bn(out2)) (padded out)
  bn_relu_nhwc_f16<true><<<(BB * HWX * 64 + 255) / 256, 256, 0, stream>>>(out2f, stats1, brg, brb, out3h);
  // 9) conv3(out3) -> f32 NCHW + stats   [WMMA]
  conv_gemm_wmma<64, 9, false, false, false, false, true, true, true>
      <<<dim3(BB * HH, 4), 256, 0, stream>>>(out3h, wpk3, 0, nullptr, nullptr, c3raw, stats2, 64);
  // 10) out4 = relu(bn(conv3))
  bn_relu_nchw_f32<<<(BB * 64 * HWX + 255) / 256, 256, 0, stream>>>(c3raw, stats2, c3g, c3b, out);
}
